// ImprovedDynamicCurvatureMLP_50611894616639
// MI455X (gfx1250) — matrix operand order repaired and checked by reading, not compiled, not run on hardware
//
#include <hip/hip_runtime.h>
#include <cstdint>
#include <cstddef>

typedef __bf16 bf16_t;
typedef __bf16 v16bf __attribute__((ext_vector_type(16)));
typedef float  v8f   __attribute__((ext_vector_type(8)));

static constexpr int BDIM = 4096, INDIM = 4096, HID = 4096, OUTD = 1024;
static constexpr float MIN_C = 1e-5f, MAX_C = 10.0f, T_GEO = 0.7f, EPSF = 1e-7f;

__device__ __forceinline__ float sigm(float x) { return 1.0f / (1.0f + __expf(-x)); }

// ---------------------------------------------------------------- converts
__global__ void cvt_bf16_kernel(const float* __restrict__ src,
                                bf16_t* __restrict__ dst, int n) {
  int i = (blockIdx.x * blockDim.x + threadIdx.x) * 4;
  if (i + 3 < n) {
    float4 v = *(const float4*)(src + i);
    union { bf16_t b[4]; uint2 u; } o;
    o.b[0] = (bf16_t)v.x; o.b[1] = (bf16_t)v.y;
    o.b[2] = (bf16_t)v.z; o.b[3] = (bf16_t)v.w;
    *(uint2*)(dst + i) = o.u;
  }
}

// W [K][N] f32 row-major  ->  Wt [N][K] bf16 row-major
__global__ void transpose_cvt_kernel(const float* __restrict__ W,
                                     bf16_t* __restrict__ Wt, int K, int N) {
  __shared__ float tile[32][33];
  int k0 = blockIdx.y * 32, n0 = blockIdx.x * 32;
  int tx = threadIdx.x, ty = threadIdx.y;        // 32 x 8
#pragma unroll
  for (int i = 0; i < 4; ++i)
    tile[ty + 8 * i][tx] = W[(size_t)(k0 + ty + 8 * i) * N + n0 + tx];
  __syncthreads();
#pragma unroll
  for (int i = 0; i < 4; ++i)
    Wt[(size_t)(n0 + ty + 8 * i) * K + k0 + tx] = (bf16_t)tile[tx][ty + 8 * i];
}

// pv[k] = sum_j proj[k][j] * curv_feat[j]
__global__ void pv_kernel(const float* __restrict__ proj,
                          const float* __restrict__ cf,
                          float* __restrict__ pv, int K) {
  int k = blockIdx.x * blockDim.x + threadIdx.x;
  if (k < K) {
    float s = 0.f;
#pragma unroll
    for (int j = 0; j < 16; ++j) s += proj[k * 16 + j] * cf[j];
    pv[k] = s;
  }
}

// ---------------------------------------------------------------- WMMA GEMM
// C = act(A[M,K] * Bt[N,K]^T + bias),  A,Bt bf16 row-major.
// Block tile 128x256x32, 8 waves in a 2x4 grid, each wave 64x64 (4x4 frags).
// Global->LDS via async-to-LDS (ASYNCcnt) with double-buffered LDS.
// EPI: 0 = tanh (write f32 + bf16), 1 = sigmoid (f32), 2 = none (f32)

static constexpr int LDS_A_BYTES = 128 * 80;           // 10240
static constexpr int LDS_B_BYTES = 256 * 80;           // 20480
static constexpr int LDS_BUF     = LDS_A_BYTES + LDS_B_BYTES;  // 30720

template <int EPI>
__global__ __launch_bounds__(256) void gemm_wmma_kernel(
    const bf16_t* __restrict__ A, const bf16_t* __restrict__ Bt,
    const float* __restrict__ bias, float* __restrict__ Cf,
    bf16_t* __restrict__ Cb, int M, int N, int K) {
  __shared__ __align__(16) uint8_t lds[2 * LDS_BUF];

  const int tid  = threadIdx.x;
  const int m0   = blockIdx.y * 128, n0 = blockIdx.x * 256;
  const int w    = tid >> 5, lane = tid & 31;
  const int wm   = w >> 2, wn = w & 3;            // 2 x 4 wave grid
  const int hl   = lane >> 4, lrow = lane & 15;   // half-wave, row-in-16

  v8f acc[4][4];
#pragma unroll
  for (int i = 0; i < 4; ++i)
#pragma unroll
    for (int j = 0; j < 4; ++j) acc[i][j] = {};

  // ---- async global->LDS staging addresses -------------------------------
  const uint32_t ldsBase = (uint32_t)(uintptr_t)(&lds[0]);
  const int rA = tid >> 1;                // 0..127: A tile row
  const int qb = (tid & 1) * 32;          // 32B chunk within 64B A row
  const int rB = tid;                     // 0..255: B tile row
  const uint32_t lA = ldsBase + (uint32_t)(rA * 80 + qb);
  const uint32_t lB = ldsBase + (uint32_t)(LDS_A_BYTES + rB * 80);
  uint32_t offA = (uint32_t)((m0 + rA) * K * 2 + qb);   // byte offsets
  uint32_t offB = (uint32_t)((n0 + rB) * K * 2);

#define ISSUE_TILE(SEL)                                                        \
  do {                                                                         \
    uint32_t la_ = lA + (SEL), lb_ = lB + (SEL);                               \
    asm volatile("global_load_async_to_lds_b128 %0, %1, %2 offset:0"           \
                 ::"v"(la_), "v"(offA), "s"(A) : "memory");                    \
    asm volatile("global_load_async_to_lds_b128 %0, %1, %2 offset:16"          \
                 ::"v"(la_), "v"(offA), "s"(A) : "memory");                    \
    asm volatile("global_load_async_to_lds_b128 %0, %1, %2 offset:0"           \
                 ::"v"(lb_), "v"(offB), "s"(Bt) : "memory");                   \
    asm volatile("global_load_async_to_lds_b128 %0, %1, %2 offset:16"          \
                 ::"v"(lb_), "v"(offB), "s"(Bt) : "memory");                   \
    asm volatile("global_load_async_to_lds_b128 %0, %1, %2 offset:32"          \
                 ::"v"(lb_), "v"(offB), "s"(Bt) : "memory");                   \
    asm volatile("global_load_async_to_lds_b128 %0, %1, %2 offset:48"          \
                 ::"v"(lb_), "v"(offB), "s"(Bt) : "memory");                   \
  } while (0)

  uint32_t sel = 0;
  ISSUE_TILE(sel);               // prologue: buffer 0, k=0
  offA += 64; offB += 64;

  for (int k0 = 0; k0 < K; k0 += 32) {
    // my previous 6 async loads (current buffer) are the only outstanding ones
    asm volatile("s_wait_asynccnt 0x0" ::: "memory");
    __syncthreads();             // everyone's current buffer is in LDS;
                                 // everyone finished reading the other buffer
    if (k0 + 32 < K) {
      uint32_t nsel = sel ^ LDS_BUF;
      ISSUE_TILE(nsel);          // overlap next tile's HBM traffic with WMMAs
      offA += 64; offB += 64;
    }

    const uint8_t* As = lds + sel;
    const uint8_t* Bs = lds + sel + LDS_A_BYTES;

    union Frag { uint4 u[2]; v16bf v; } af, bg[4];
#pragma unroll
    for (int j = 0; j < 4; ++j) {           // B: contiguous 16-K per half
      const uint8_t* p = Bs + (wn * 64 + j * 16 + lrow) * 80 + hl * 32;
      bg[j].u[0] = *(const uint4*)p;
      bg[j].u[1] = *(const uint4*)(p + 16);
    }
#pragma unroll
    for (int i = 0; i < 4; ++i) {           // A: contiguous 16-K per half
      const uint8_t* p = As + (wm * 64 + i * 16 + lrow) * 80 + hl * 32;
      af.u[0] = *(const uint4*)p;
      af.u[1] = *(const uint4*)(p + 16);
#pragma unroll
      for (int j = 0; j < 4; ++j)
        acc[i][j] = __builtin_amdgcn_wmma_f32_16x16x32_bf16(
            false, af.v, false, bg[j].v, (short)0, acc[i][j], false, false);
    }
    sel ^= LDS_BUF;
  }
#undef ISSUE_TILE

#pragma unroll
  for (int i = 0; i < 4; ++i) {
    const int mb = m0 + wm * 64 + i * 16 + hl * 8;
#pragma unroll
    for (int j = 0; j < 4; ++j) {
      const int n = n0 + wn * 64 + j * 16 + lrow;
      const float bv = bias[n];
#pragma unroll
      for (int rr = 0; rr < 8; ++rr) {
        float v = acc[i][j][rr] + bv;
        if (EPI == 0)      v = tanhf(v);
        else if (EPI == 1) v = sigm(v);
        const size_t idx = (size_t)(mb + rr) * N + n;
        Cf[idx] = v;
        if (EPI == 0) Cb[idx] = (bf16_t)v;
      }
    }
  }
}

// ---------------------------------------------------------------- row stats
__global__ void rowstats_kernel(const float* __restrict__ Hf,
                                const float* __restrict__ Uf,
                                const float* __restrict__ x,
                                const float* __restrict__ pv,
                                const float* __restrict__ curv_scale,
                                float2* __restrict__ coef, int D, int IND) {
  const int b = blockIdx.x, tid = threadIdx.x;
  const float* hr = Hf + (size_t)b * D;
  const float* ur = Uf + (size_t)b * D;
  const float* xr = x + (size_t)b * IND;
  float hh = 0.f, uu = 0.f, hu = 0.f, fe = 0.f;
  for (int k = tid; k < D; k += 256) {
    float h = hr[k], u = ur[k];
    hh += h * h; uu += u * u; hu += h * u;
  }
  for (int k = tid; k < IND; k += 256) fe += xr[k] * pv[k];
  __shared__ float s[4][256];
  s[0][tid] = hh; s[1][tid] = uu; s[2][tid] = hu; s[3][tid] = fe;
  __syncthreads();
  for (int st = 128; st > 0; st >>= 1) {
    if (tid < st)
#pragma unroll
      for (int q = 0; q < 4; ++q) s[q][tid] += s[q][tid + st];
    __syncthreads();
  }
  if (tid == 0) {
    hh = s[0][0]; uu = s[1][0]; hu = s[2][0]; fe = s[3][0];
    float c = MIN_C + (MAX_C - MIN_C) * sigm(fe + curv_scale[0]);
    c = fminf(fmaxf(c, MIN_C), MAX_C);
    const float sc = sqrtf(c);
    const float nh = fmaxf(sqrtf(hh), EPSF);
    const float ah = fminf(fmaxf(sc * nh, EPSF), 1.0f - 1e-5f);
    const float sa = tanhf((1.0f - T_GEO) * 0.5f * log1pf(2.0f * ah / (1.0f - ah)))
                     / (sc * nh);
    const float nu = fmaxf(sqrtf(uu), EPSF);
    const float au = fminf(fmaxf(sc * nu, EPSF), 1.0f - 1e-5f);
    const float sb = tanhf(T_GEO * 0.5f * log1pf(2.0f * au / (1.0f - au)))
                     / (sc * nu);
    const float a2 = sa * sa * hh, b2 = sb * sb * uu, ab = sa * sb * hu;
    const float na = 1.0f + 2.0f * c * ab + c * b2;
    const float nb = 1.0f - c * a2;
    float den = 1.0f + 2.0f * c * ab + c * c * a2 * b2;
    den = fmaxf(den, EPSF);
    float alpha = na * sa / den, beta = nb * sb / den;
    const bool bad = !(alpha == alpha) || !(beta == beta) ||
                     fabsf(alpha) > 3.0e38f || fabsf(beta) > 3.0e38f;
    if (bad) { alpha = 1.0f; beta = 0.0f; }   // z row falls back to h
    coef[b] = make_float2(alpha, beta);
  }
}

// z = alpha*h + beta*u  -> bf16
__global__ void combine_kernel(const float* __restrict__ Hf,
                               const float* __restrict__ Uf,
                               const float2* __restrict__ coef,
                               bf16_t* __restrict__ Zb, int D) {
  size_t i = ((size_t)blockIdx.x * blockDim.x + threadIdx.x) * 4;
  const float2 ab = coef[i / (size_t)D];
  float4 h = *(const float4*)(Hf + i);
  float4 u = *(const float4*)(Uf + i);
  union { bf16_t b[4]; uint2 u2; } o;
  o.b[0] = (bf16_t)(ab.x * h.x + ab.y * u.x);
  o.b[1] = (bf16_t)(ab.x * h.y + ab.y * u.y);
  o.b[2] = (bf16_t)(ab.x * h.z + ab.y * u.z);
  o.b[3] = (bf16_t)(ab.x * h.w + ab.y * u.w);
  *(uint2*)(Zb + i) = o.u2;
}

// ---------------------------------------------------------------- launch
extern "C" void kernel_launch(void* const* d_in, const int* in_sizes, int n_in,
                              void* d_out, int out_size, void* d_ws,
                              size_t ws_size, hipStream_t stream) {
  const float* x    = (const float*)d_in[0];
  const float* W1   = (const float*)d_in[1];
  const float* b1   = (const float*)d_in[2];
  const float* W2   = (const float*)d_in[3];
  const float* b2   = (const float*)d_in[4];
  const float* Wout = (const float*)d_in[5];
  const float* bout = (const float*)d_in[6];
  const float* cfeat = (const float*)d_in[7];
  const float* cscale = (const float*)d_in[8];
  const float* proj = (const float*)d_in[9];
  float* out = (float*)d_out;

  uint8_t* ws = (uint8_t*)d_ws;
  const size_t MB = 1024ull * 1024ull;
  bf16_t* Xb    = (bf16_t*)(ws + 0);           // 32 MB
  bf16_t* W1t   = (bf16_t*)(ws + 32 * MB);     // 32 MB
  bf16_t* W2t   = (bf16_t*)(ws + 64 * MB);     // 32 MB
  bf16_t* Woutt = (bf16_t*)(ws + 96 * MB);     // 8 MB
  float*  Hf    = (float*)(ws + 104 * MB);     // 64 MB
  bf16_t* Hb    = (bf16_t*)(ws + 168 * MB);    // 32 MB
  float*  Uf    = (float*)(ws + 200 * MB);     // 64 MB
  bf16_t* Zb    = (bf16_t*)(ws + 264 * MB);    // 32 MB
  float2* coef  = (float2*)(ws + 296 * MB);    // 32 KB
  float*  pv    = (float*)(ws + 296 * MB + 64 * 1024);

  // bf16 staging
  {
    int n = BDIM * INDIM;
    cvt_bf16_kernel<<<n / 1024, 256, 0, stream>>>(x, Xb, n);
  }
  transpose_cvt_kernel<<<dim3(HID / 32, INDIM / 32), dim3(32, 8), 0, stream>>>(
      W1, W1t, INDIM, HID);
  transpose_cvt_kernel<<<dim3(HID / 32, HID / 32), dim3(32, 8), 0, stream>>>(
      W2, W2t, HID, HID);
  transpose_cvt_kernel<<<dim3(OUTD / 32, HID / 32), dim3(32, 8), 0, stream>>>(
      Wout, Woutt, HID, OUTD);
  pv_kernel<<<INDIM / 256, 256, 0, stream>>>(proj, cfeat, pv, INDIM);

  // h = tanh(x W1 + b1)
  gemm_wmma_kernel<0><<<dim3(HID / 256, BDIM / 128), 256, 0, stream>>>(
      Xb, W1t, b1, Hf, Hb, BDIM, HID, INDIM);
  // u = sigmoid(h W2 + b2)
  gemm_wmma_kernel<1><<<dim3(HID / 256, BDIM / 128), 256, 0, stream>>>(
      Hb, W2t, b2, Uf, nullptr, BDIM, HID, HID);
  // per-row poincare coefficients
  rowstats_kernel<<<BDIM, 256, 0, stream>>>(Hf, Uf, x, pv, cscale, coef, HID,
                                            INDIM);
  // z = alpha h + beta u
  combine_kernel<<<(BDIM * HID / 4) / 256, 256, 0, stream>>>(Hf, Uf, coef, Zb,
                                                             HID);
  // out = z Wout + bout
  gemm_wmma_kernel<2><<<dim3(OUTD / 256, BDIM / 128), 256, 0, stream>>>(
      Zb, Woutt, bout, out, nullptr, BDIM, OUTD, HID);
}
